// MultiHead_11347303596636
// MI455X (gfx1250) — matrix operand order repaired and checked by reading, not compiled, not run on hardware
//
#include <hip/hip_runtime.h>

typedef __bf16 bf16;
typedef __attribute__((ext_vector_type(16))) __bf16 v16bf;
typedef __attribute__((ext_vector_type(8)))  __bf16 v8bf;
typedef __attribute__((ext_vector_type(4)))  __bf16 v4bf;
typedef __attribute__((ext_vector_type(8)))  float  v8f;
typedef __attribute__((ext_vector_type(4)))  unsigned int u32x4;
typedef __attribute__((ext_vector_type(8)))  int i32x8;
typedef __attribute__((ext_vector_type(4)))  int i32x4;

#define DIM   1024
#define NH    16
#define HD    64
#define SEQ   2048
#define BATCH 2
#define MTOT  (BATCH * SEQ)   // 4096

// ---------------------------------------------------------------- helpers
__device__ __forceinline__ v16bf cat16(v8bf lo, v8bf hi) {
  return __builtin_shufflevector(lo, hi, 0,1,2,3,4,5,6,7,8,9,10,11,12,13,14,15);
}
__device__ __forceinline__ v8bf ldg8(const bf16* p) { return *(const v8bf*)p; }

__device__ __forceinline__ v8f wmma_bf16(v16bf a, v16bf b, v8f c) {
  // D(16x16,f32) = A(16x32,bf16) * B(32x16,bf16) + C
  return __builtin_amdgcn_wmma_f32_16x16x32_bf16(false, a, false, b, (short)0, c,
                                                 false, false);
}
__device__ __forceinline__ v8f v8f_zero() {
  v8f z = {0.f,0.f,0.f,0.f,0.f,0.f,0.f,0.f};
  return z;
}

// TDM: DMA a 2D bf16 tile [tile1 rows x tile0 elems] from global into LDS.
// Builds D# group0/group1 per CDNA5 ISA 8.3/8.4; groups 2/3 zero (2D tensor).
// Toolchain uses the 6-arg builtin form (extra i32x8 before cpol).
__device__ __forceinline__ void tdm_load_2d(unsigned lds_off, const bf16* gptr,
                                            unsigned tdim0, unsigned tdim1,
                                            unsigned tile0, unsigned tile1,
                                            unsigned stride0) {
  unsigned long long ga = (unsigned long long)(uintptr_t)gptr;
  u32x4 g0;
  g0[0] = 1u;                                                // count=1, no gather
  g0[1] = lds_off;                                           // lds_addr (bytes)
  g0[2] = (unsigned)ga;                                      // global_addr[31:0]
  g0[3] = (unsigned)((ga >> 32) & 0x01ffffffu) | (2u << 30); // ga[56:32] | type=2
  i32x8 g1;
  g1[0] = (int)(1u << 16);                                   // wg_mask=0, data_size=1 (2B)
  g1[1] = (int)((tdim0 & 0xffffu) << 16);                    // tensor_dim0[15:0]
  g1[2] = (int)(((tdim0 >> 16) & 0xffffu) | ((tdim1 & 0xffffu) << 16));
  g1[3] = (int)(((tdim1 >> 16) & 0xffffu) | (tile0 << 16));  // tile_dim0
  g1[4] = (int)tile1;                                        // tile_dim1 (tile_dim2=0)
  g1[5] = (int)stride0;                                      // tensor_dim0_stride[31:0]
  g1[6] = 0;
  g1[7] = 0;
  i32x4 z4 = {0, 0, 0, 0};
  i32x8 z8 = {0, 0, 0, 0, 0, 0, 0, 0};
  __builtin_amdgcn_tensor_load_to_lds(g0, g1, z4, z4, z8, 0);
}

// ---------------------------------------------------------------- fp32 -> bf16
__global__ void cvt_bf16_kernel(const float* __restrict__ src,
                                bf16* __restrict__ dst, int n) {
  int i = (blockIdx.x * blockDim.x + threadIdx.x) * 4;
  if (i < n) {
    const float4 f = *(const float4*)(src + i);
    v4bf o;
    o[0] = (bf16)f.x; o[1] = (bf16)f.y; o[2] = (bf16)f.z; o[3] = (bf16)f.w;
    *(v4bf*)(dst + i) = o;
  }
}

// ---------------------------------------------------------------- GEMM  Y = X * W^T
// X: [M, DIM] bf16 row-major, W: [DIM, DIM] bf16 row-major (Linear weight).
// mode 0: store bf16 head-major   [B,NH,SEQ,HD]   (Q, K)
// mode 2: store bf16 head-major transposed [B,NH,HD,SEQ]  (V)
// mode 1: store fp32 row-major    [M, DIM]        (final output)
// WG = 128 thr = 4 waves in 2x2; wave tile 64x64 (4x4 WMMA tiles).
// A/W tiles (128x32 each) staged into LDS by the Tensor Data Mover.
__global__ __launch_bounds__(128)
void gemm_bf16_kernel(const bf16* __restrict__ A, const bf16* __restrict__ W,
                      bf16* __restrict__ outb, float* __restrict__ outf, int mode) {
  const int lane = threadIdx.x & 31;
  const int wid  = threadIdx.x >> 5;
  const int hi   = lane >> 4;      // half-wave select
  const int ln   = lane & 15;
  const int wm = wid >> 1, wn = wid & 1;
  const int mblk = blockIdx.x * 128;
  const int nblk = blockIdx.y * 128;

  __shared__ __align__(16) bf16 lds_a[128 * 32];   // A tile: rows x 32-K chunk
  __shared__ __align__(16) bf16 lds_w[128 * 32];   // W tile
  const unsigned lds_a_off = (unsigned)(uintptr_t)(void*)lds_a;
  const unsigned lds_w_off = (unsigned)(uintptr_t)(void*)lds_w;

  v8f acc[4][4];
#pragma unroll
  for (int i = 0; i < 4; ++i)
#pragma unroll
    for (int j = 0; j < 4; ++j) acc[i][j] = v8f_zero();

  for (int kk = 0; kk < DIM; kk += 32) {
    __syncthreads();                       // previous tile fully consumed
    if (wid == 0) {
      tdm_load_2d(lds_a_off, A + (size_t)mblk * DIM + kk, DIM, MTOT, 32, 128, DIM);
      tdm_load_2d(lds_w_off, W + (size_t)nblk * DIM + kk, DIM, DIM,  32, 128, DIM);
      __builtin_amdgcn_s_wait_tensorcnt(0);
    }
    __syncthreads();                       // tile visible to all waves

    v16bf af[4], bfr[4];
#pragma unroll
    for (int mt = 0; mt < 4; ++mt) {
      // A-layout: lane<16 -> K{0..7,16..23}; lane>=16 -> K{8..15,24..31}
      const bf16* pa = lds_a + (wm * 64 + 16 * mt + ln) * 32 + hi * 8;
      af[mt] = cat16(*(const v8bf*)pa, *(const v8bf*)(pa + 16));
    }
#pragma unroll
    for (int nt = 0; nt < 4; ++nt) {
      // B-layout: lane<16 -> K{0..7,16..23}; lane>=16 -> K{8..15,24..31} (col = row of W)
      const bf16* pb = lds_w + (wn * 64 + 16 * nt + ln) * 32 + hi * 8;
      bfr[nt] = cat16(*(const v8bf*)pb, *(const v8bf*)(pb + 16));
    }
#pragma unroll
    for (int mt = 0; mt < 4; ++mt)
#pragma unroll
      for (int nt = 0; nt < 4; ++nt)
        acc[mt][nt] = wmma_bf16(af[mt], bfr[nt], acc[mt][nt]);
  }

  // C layout: lanes 0-15 vgpr r -> (M=r, N=ln); lanes 16-31 -> (M=r+8, N=ln)
#pragma unroll
  for (int mt = 0; mt < 4; ++mt)
#pragma unroll
    for (int nt = 0; nt < 4; ++nt)
#pragma unroll
      for (int r = 0; r < 8; ++r) {
        const int row = mblk + wm * 64 + 16 * mt + r + 8 * hi;
        const int col = nblk + wn * 64 + 16 * nt + ln;
        const float val = acc[mt][nt][r];
        if (mode == 1) {
          outf[(size_t)row * DIM + col] = val;
        } else {
          const int b = row >> 11, s = row & (SEQ - 1);
          const int h = col >> 6,  d = col & (HD - 1);
          size_t idx;
          if (mode == 0) idx = (((size_t)(b * NH + h) * SEQ) + s) * HD + d;
          else           idx = (((size_t)(b * NH + h) * HD)  + d) * SEQ + s;
          outb[idx] = (bf16)val;
        }
      }
}

// ---------------------------------------------------------------- causal flash attention
// Qh,Kh: [B,NH,SEQ,HD] bf16;  Vt: [B,NH,HD,SEQ] bf16;  Ao: [B,SEQ,DIM] bf16
// 4 waves / WG; each wave owns a 16-query tile, streams 32-key blocks.
__global__ __launch_bounds__(128)
void attn_kernel(const bf16* __restrict__ Qh, const bf16* __restrict__ Kh,
                 const bf16* __restrict__ Vt, bf16* __restrict__ Ao) {
  const int lane = threadIdx.x & 31;
  const int wid  = threadIdx.x >> 5;
  const int hi = lane >> 4, ln = lane & 15;
  const int bh = blockIdx.y;
  const int q0 = blockIdx.x * 64 + wid * 16;

  const bf16* Qb = Qh + (size_t)bh * SEQ * HD;
  const bf16* Kb = Kh + (size_t)bh * SEQ * HD;
  const bf16* Vb = Vt + (size_t)bh * HD * SEQ;

  __shared__ __align__(16) bf16 lds_p[4][16 * 32];  // per-wave P tile (bf16)
  bf16* lp = lds_p[wid];

  // Q fragments (16 rows x 64 d) -> two A-frags (K=32 each), loaded once
  v16bf qf[2];
  {
    const bf16* p = Qb + (size_t)(q0 + ln) * HD;
#pragma unroll
    for (int db = 0; db < 2; ++db) {
      const bf16* pp = p + db * 32 + hi * 8;
      qf[db] = cat16(ldg8(pp), ldg8(pp + 16));
    }
  }

  v8f o[4];
#pragma unroll
  for (int t = 0; t < 4; ++t) o[t] = v8f_zero();
  float rm[8], rl[8];
#pragma unroll
  for (int r = 0; r < 8; ++r) { rm[r] = -1e30f; rl[r] = 0.f; }

  const float sc = 0.125f;  // 1/sqrt(HD)

  for (int kb = 0; kb < q0 + 16; kb += 32) {
    // hint: pull the next key block toward the WGP while we compute this one
    __builtin_prefetch(Kb + (size_t)(kb + 32 + ln) * HD, 0, 1);

    // ---- S = Q K^T for 32 keys (two 16-key column groups)
    v8f s0 = v8f_zero(), s1 = v8f_zero();
    {
      const bf16* pk0 = Kb + (size_t)(kb + ln) * HD + hi * 8;
      s0 = wmma_bf16(qf[0], cat16(ldg8(pk0),      ldg8(pk0 + 16)),     s0);
      s0 = wmma_bf16(qf[1], cat16(ldg8(pk0 + 32), ldg8(pk0 + 48)),     s0);
      const bf16* pk1 = Kb + (size_t)(kb + 16 + ln) * HD + hi * 8;
      s1 = wmma_bf16(qf[0], cat16(ldg8(pk1),      ldg8(pk1 + 16)),     s1);
      s1 = wmma_bf16(qf[1], cat16(ldg8(pk1 + 32), ldg8(pk1 + 48)),     s1);
    }

    // ---- scale, causal mask, online softmax
    float p0a[8], p1a[8], alpha[8];
#pragma unroll
    for (int r = 0; r < 8; ++r) {
      const int qrow = q0 + r + 8 * hi;
      float v0 = s0[r] * sc;
      float v1 = s1[r] * sc;
      if (kb + ln > qrow)      v0 = -1e30f;
      if (kb + 16 + ln > qrow) v1 = -1e30f;
      float mx = fmaxf(v0, v1);
#pragma unroll
      for (int off = 1; off < 16; off <<= 1)
        mx = fmaxf(mx, __shfl_xor(mx, off, 32));   // reduce within 16-lane half
      const float mnew = fmaxf(rm[r], mx);
      const float a = __expf(rm[r] - mnew);
      rm[r] = mnew; alpha[r] = a;
      const float p0 = __expf(v0 - mnew);
      const float p1 = __expf(v1 - mnew);
      float rs = p0 + p1;
#pragma unroll
      for (int off = 1; off < 16; off <<= 1)
        rs += __shfl_xor(rs, off, 32);
      rl[r] = rl[r] * a + rs;
      p0a[r] = p0; p1a[r] = p1;
    }
#pragma unroll
    for (int t = 0; t < 4; ++t)
#pragma unroll
      for (int r = 0; r < 8; ++r) o[t][r] *= alpha[r];

    // ---- re-swizzle P (C-layout f32) -> A-layout bf16 via per-wave LDS
#pragma unroll
    for (int r = 0; r < 8; ++r) {
      const int row = r + 8 * hi;
      lp[row * 32 + ln]      = (bf16)p0a[r];
      lp[row * 32 + 16 + ln] = (bf16)p1a[r];
    }
    asm volatile("s_wait_dscnt 0" ::: "memory");  // lane-crossing LDS visibility (wave-local)
    v16bf pf;
    {
      const bf16* pp = lp + ln * 32 + hi * 8;
      pf = cat16(*(const v8bf*)pp, *(const v8bf*)(pp + 16));
    }

    // ---- O += P * V  (V pre-transposed -> contiguous B-frag loads)
#pragma unroll
    for (int t = 0; t < 4; ++t) {
      const bf16* pv = Vb + (size_t)(16 * t + ln) * SEQ + kb + hi * 8;
      o[t] = wmma_bf16(pf, cat16(ldg8(pv), ldg8(pv + 16)), o[t]);
    }
  }

  // ---- epilogue: normalize, write token-major bf16 for the Wo GEMM
  const int b = bh >> 4, h = bh & 15;
#pragma unroll
  for (int t = 0; t < 4; ++t)
#pragma unroll
    for (int r = 0; r < 8; ++r) {
      const int qrow = q0 + r + 8 * hi;
      const float val = o[t][r] / rl[r];
      Ao[((size_t)(b * SEQ + qrow)) * DIM + h * HD + 16 * t + ln] = (bf16)val;
    }
}

// ---------------------------------------------------------------- launch
extern "C" void kernel_launch(void* const* d_in, const int* in_sizes, int n_in,
                              void* d_out, int out_size, void* d_ws, size_t ws_size,
                              hipStream_t stream) {
  (void)in_sizes; (void)n_in; (void)out_size; (void)ws_size;
  const float* q  = (const float*)d_in[0];
  const float* k  = (const float*)d_in[1];
  const float* v  = (const float*)d_in[2];
  const float* wq = (const float*)d_in[3];
  const float* wk = (const float*)d_in[4];
  const float* wv = (const float*)d_in[5];
  const float* wo = (const float*)d_in[6];
  float* out = (float*)d_out;

  const size_t BSD = (size_t)BATCH * SEQ * DIM;  // 4,194,304
  const size_t DD  = (size_t)DIM * DIM;          // 1,048,576

  bf16* w = (bf16*)d_ws;
  bf16* qb  = w; w += BSD;
  bf16* kb  = w; w += BSD;
  bf16* vb  = w; w += BSD;
  bf16* wqb = w; w += DD;
  bf16* wkb = w; w += DD;
  bf16* wvb = w; w += DD;
  bf16* wob = w; w += DD;
  bf16* Qh  = w; w += BSD;
  bf16* Kh  = w; w += BSD;
  bf16* Vt  = w; w += BSD;
  bf16* Ao  = qb;  // qb is dead after its GEMM; alias to save workspace

  const int cvt_blk = 256;
  cvt_bf16_kernel<<<(int)(BSD / 4 + cvt_blk - 1) / cvt_blk, cvt_blk, 0, stream>>>(q,  qb,  (int)BSD);
  cvt_bf16_kernel<<<(int)(BSD / 4 + cvt_blk - 1) / cvt_blk, cvt_blk, 0, stream>>>(k,  kb,  (int)BSD);
  cvt_bf16_kernel<<<(int)(BSD / 4 + cvt_blk - 1) / cvt_blk, cvt_blk, 0, stream>>>(v,  vb,  (int)BSD);
  cvt_bf16_kernel<<<(int)(DD  / 4 + cvt_blk - 1) / cvt_blk, cvt_blk, 0, stream>>>(wq, wqb, (int)DD);
  cvt_bf16_kernel<<<(int)(DD  / 4 + cvt_blk - 1) / cvt_blk, cvt_blk, 0, stream>>>(wk, wkb, (int)DD);
  cvt_bf16_kernel<<<(int)(DD  / 4 + cvt_blk - 1) / cvt_blk, cvt_blk, 0, stream>>>(wv, wvb, (int)DD);
  cvt_bf16_kernel<<<(int)(DD  / 4 + cvt_blk - 1) / cvt_blk, cvt_blk, 0, stream>>>(wo, wob, (int)DD);

  dim3 gg(MTOT / 128, DIM / 128);  // 32 x 8 workgroups, 4 waves each
  gemm_bf16_kernel<<<gg, 128, 0, stream>>>(qb, wqb, Qh, nullptr, 0);
  gemm_bf16_kernel<<<gg, 128, 0, stream>>>(kb, wkb, Kh, nullptr, 0);
  gemm_bf16_kernel<<<gg, 128, 0, stream>>>(vb, wvb, Vt, nullptr, 2);

  attn_kernel<<<dim3(SEQ / 64, BATCH * NH), 128, 0, stream>>>(Qh, Kh, Vt, Ao);

  gemm_bf16_kernel<<<gg, 128, 0, stream>>>(Ao, wob, nullptr, out, 1);
}
